// ExpertsLayer_6966436954205
// MI455X (gfx1250) — matrix operand order repaired and checked by reading, not compiled, not run on hardware
//
#include <hip/hip_runtime.h>

#define D_ 256
#define H_ 1024
#define E_ 8
#define T_ 65536
#define CAP 10240
#define NTILE (CAP / 64)   // 160 tiles of 64 slots per expert

typedef __attribute__((ext_vector_type(16))) __bf16 v16bf;
typedef __attribute__((ext_vector_type(8)))  float  v8f;

// ---------- helpers ----------

static __device__ __forceinline__ unsigned short f2bf(float f) {
    unsigned u = __float_as_uint(f);
    unsigned r = u + 0x7FFFu + ((u >> 16) & 1u);   // round-to-nearest-even
    return (unsigned short)(r >> 16);
}
static __device__ __forceinline__ unsigned pack2bf(float a, float b) {
    return (unsigned)f2bf(a) | ((unsigned)f2bf(b) << 16);
}

union FragU { uint4 q[2]; v16bf v; };

// 16 contiguous bf16 (two 16B chunks) -> fragment (B-matrix layout)
static __device__ __forceinline__ v16bf ldfrag1(const unsigned short* p) {
    FragU u;
    u.q[0] = *(const uint4*)(p);
    u.q[1] = *(const uint4*)(p + 8);
    return u.v;
}
// two separated 8-bf16 chunks -> fragment (A-matrix layout)
static __device__ __forceinline__ v16bf ldfrag2(const unsigned short* p0,
                                                const unsigned short* p1) {
    FragU u;
    u.q[0] = *(const uint4*)(p0);
    u.q[1] = *(const uint4*)(p1);
    return u.v;
}

// CDNA5 async global->LDS copy, 16B per lane (tracked with ASYNCcnt)
static __device__ __forceinline__ void async_copy_b128(unsigned lds_off,
                                                       const void* gaddr) {
    asm volatile("global_load_async_to_lds_b128 %0, %1, off"
                 :
                 : "v"(lds_off), "v"(gaddr)
                 : "memory");
}
static __device__ __forceinline__ void wait_async0() {
    asm volatile("s_wait_asynccnt 0x0" ::: "memory");
}

// ---------- K0: weight conversion (packed) + inv_idx init ----------

__global__ void moe_prep_kernel(const float* __restrict__ W1,
                                const float* __restrict__ W2,
                                unsigned* __restrict__ w1bf32,
                                unsigned* __restrict__ w2bf32,
                                int* __restrict__ inv_idx) {
    size_t i = (size_t)blockIdx.x * blockDim.x + threadIdx.x;
    size_t stride = (size_t)gridDim.x * blockDim.x;
    const size_t NW2 = (size_t)E_ * D_ * H_ / 2;     // 1M packed pairs each
    for (size_t j = i; j < NW2; j += stride) {
        w1bf32[j] = pack2bf(W1[2 * j], W1[2 * j + 1]);
        w2bf32[j] = pack2bf(W2[2 * j], W2[2 * j + 1]);
    }
    for (size_t j = i; j < (size_t)E_ * CAP; j += stride) inv_idx[j] = -1;
}

// ---------- K1: router (softmax, top-1, deterministic block-local ranks) ----------

__global__ __launch_bounds__(256) void moe_router_kernel(
        const float* __restrict__ x, const float* __restrict__ Wg,
        const float* __restrict__ bg,
        int* __restrict__ routeArr, int* __restrict__ rankArr,
        int* __restrict__ blockhist, float* __restrict__ wavepart,
        float* __restrict__ pmOut) {
    __shared__ float sWg[D_ * E_];
    __shared__ float sbg[E_];
    __shared__ int   sroute[256];
    __shared__ int   srank[256];

    const int tid = threadIdx.x, lane = tid & 31, wave = tid >> 5;
    const int bx = blockIdx.x;

    for (int i = tid; i < D_ * E_; i += 256) sWg[i] = Wg[i];
    if (tid < E_) sbg[tid] = bg[tid];
    __syncthreads();

    float wacc[E_];
#pragma unroll
    for (int e = 0; e < E_; e++) wacc[e] = 0.f;

    const int tokBase = bx * 256 + wave * 32;
    for (int i = 0; i < 32; i++) {
        const int t = tokBase + i;
        const float* xr = x + (size_t)t * D_;
        float acc[E_];
#pragma unroll
        for (int e = 0; e < E_; e++) acc[e] = 0.f;
#pragma unroll
        for (int ii = 0; ii < 8; ii++) {
            int d = lane + ii * 32;
            float xv = xr[d];
#pragma unroll
            for (int e = 0; e < E_; e++) acc[e] += xv * sWg[d * E_ + e];
        }
#pragma unroll
        for (int off = 16; off >= 1; off >>= 1) {
#pragma unroll
            for (int e = 0; e < E_; e++) acc[e] += __shfl_xor(acc[e], off, 32);
        }
        float l[E_];
#pragma unroll
        for (int e = 0; e < E_; e++) l[e] = acc[e] + sbg[e];
        float m = l[0]; int arg = 0;
#pragma unroll
        for (int e = 1; e < E_; e++) { if (l[e] > m) { m = l[e]; arg = e; } }
        float p[E_]; float s = 0.f;
#pragma unroll
        for (int e = 0; e < E_; e++) { p[e] = __expf(l[e] - m); s += p[e]; }
        float inv = 1.f / s;
#pragma unroll
        for (int e = 0; e < E_; e++) wacc[e] += p[e] * inv;
        if (lane == 0) {
            sroute[wave * 32 + i] = arg;
            pmOut[t] = inv;              // p[arg] == 1 -> prob_max = 1/s
        }
    }
    if (lane == 0) {
#pragma unroll
        for (int e = 0; e < E_; e++) wavepart[(bx * 8 + wave) * E_ + e] = wacc[e];
    }
    __syncthreads();
    if (tid < E_) {                       // deterministic token-order rank
        int e = tid, cnt = 0;
        for (int i = 0; i < 256; i++)
            if (sroute[i] == e) srank[i] = cnt++;
        blockhist[bx * E_ + e] = cnt;
    }
    __syncthreads();
    routeArr[bx * 256 + tid] = sroute[tid];
    rankArr[bx * 256 + tid]  = srank[tid];
}

// ---------- K2: exclusive scan over blocks + small reductions ----------

__global__ void moe_scan_kernel(const int* __restrict__ blockhist,
                                const float* __restrict__ wavepart,
                                int* __restrict__ blockoff,
                                float* __restrict__ out_counts,
                                float* __restrict__ out_psum,
                                float* __restrict__ out_ndrop) {
    __shared__ int scnt[E_];
    int tid = threadIdx.x;
    if (tid < E_) {
        int e = tid, off = 0;
        for (int b = 0; b < 256; b++) {
            blockoff[b * E_ + e] = off;
            off += blockhist[b * E_ + e];
        }
        scnt[e] = off;
        out_counts[e] = (float)off;
        float s = 0.f;
        for (int w = 0; w < 2048; w++) s += wavepart[w * E_ + e];
        out_psum[e] = s;
    }
    __syncthreads();
    if (tid == 0) {
        int kept = 0;
        for (int e = 0; e < E_; e++) kept += (scnt[e] < CAP) ? scnt[e] : CAP;
        out_ndrop[0] = (float)(T_ - kept);
    }
}

// ---------- K3: dispatch (gather kept tokens -> bf16 buffer; dropped -> y) ----------

__global__ __launch_bounds__(256) void moe_dispatch_kernel(
        const float* __restrict__ x, const int* __restrict__ routeArr,
        const int* __restrict__ rankArr, const int* __restrict__ blockoff,
        const float* __restrict__ pmIn, unsigned short* __restrict__ bufbf,
        int* __restrict__ inv_idx, float* __restrict__ y) {
    const int tid = threadIdx.x, lane = tid & 31, wave = tid >> 5;
    const int t = blockIdx.x * 8 + wave;
    const int e = routeArr[t];
    const int pos = blockoff[(t >> 8) * E_ + e] + rankArr[t];
    const float* xr = x + (size_t)t * D_;
    if (pos < CAP) {
        unsigned* br32 = (unsigned*)(bufbf + ((size_t)(e * CAP + pos)) * D_);
#pragma unroll
        for (int i = 0; i < 4; i++) {
            float2 xv = *(const float2*)(xr + lane * 2 + i * 64);
            br32[lane + i * 32] = pack2bf(xv.x, xv.y);
        }
        if (lane == 0) inv_idx[e * CAP + pos] = t;
    } else {
        float pmv = pmIn[t];
        float2* yr = (float2*)(y + (size_t)t * D_);
#pragma unroll
        for (int i = 0; i < 4; i++) {
            float2 xv = *(const float2*)(xr + lane * 2 + i * 64);
            float2 ov; ov.x = xv.x * pmv; ov.y = xv.y * pmv;
            yr[lane + i * 32] = ov;
        }
    }
}

// ---------- K4: fused expert MLP (fc1 -> relu -> fc2) via WMMA bf16 ----------
// One workgroup = 64 capacity slots of one expert. 8 waves (wave32).
// GEMM1: H-tile 64x64 per h-block, waves 4(M) x 2(N), 2 accum tiles each.
// GEMM2: eo 64x256 held in registers, waves 4(M) x 2(N/2), 8 accum tiles each.

__global__ __launch_bounds__(256) void moe_expert_kernel(
        const unsigned short* __restrict__ bufbf,  // [E][CAP][D] bf16
        const unsigned short* __restrict__ w1bf,   // [E][D][H]  bf16
        const unsigned short* __restrict__ w2bf,   // [E][H][D]  bf16
        const float* __restrict__ b1,              // [E][H]
        const float* __restrict__ b2,              // [E][D]
        const int* __restrict__ inv_idx,           // [E][CAP]
        const float* __restrict__ pm,              // [T]
        float* __restrict__ y) {                   // [T][D]
    __shared__ __align__(16) unsigned short As[64 * 264];   // A tile, padded
    __shared__ __align__(16) unsigned short Bs[256 * 40];   // W slab, transposed [n][k], padded
    __shared__ __align__(16) unsigned short Hs[64 * 72];    // relu(h) tile, padded

    const int e = blockIdx.x / NTILE;
    const int tile0 = (blockIdx.x % NTILE) * 64;
    const int tid = threadIdx.x;
    const int lane = tid & 31;
    const int wave = tid >> 5;
    const int hl = lane & 15;      // half-lane index (N / M-row selector)
    const int hi = lane >> 4;      // 0: lanes 0-15, 1: lanes 16-31
    const int mi = wave & 3;       // M tile row (rows mi*16 .. +16)
    const int nh = wave >> 2;      // N segment

    // Stage A tile (64 slots x 256 bf16 = 32KB, contiguous) via async DMA to LDS.
    // 2048 x 16B chunks; 8 per thread; LDS rows are padded to 264 bf16.
    {
        const unsigned as_base = (unsigned)(uintptr_t)(&As[0]);
        const unsigned short* gbase =
            bufbf + ((size_t)(e * CAP + tile0)) * D_;
#pragma unroll
        for (int j = 0; j < 8; j++) {
            int c = j * 256 + tid;                 // chunk id
            int row = c >> 5;                      // 32 chunks per 256-elem row
            int col = c & 31;
            unsigned lds_off = as_base + (unsigned)(row * 528 + col * 16);
            const void* g = (const void*)(gbase + (size_t)row * 256 + col * 8);
            async_copy_b128(lds_off, g);
        }
        wait_async0();
    }
    __syncthreads();

    const v8f vzero = {0.f, 0.f, 0.f, 0.f, 0.f, 0.f, 0.f, 0.f};
    v8f eo[8];
#pragma unroll
    for (int i = 0; i < 8; i++) eo[i] = vzero;

    unsigned* Bs32 = (unsigned*)Bs;                // u32 view: index n*20 + k/2

    for (int hb = 0; hb < H_; hb += 64) {
        // ---------------- GEMM1: H-tile = relu(A @ W1[:, hb:hb+64] + b1) ----------------
        v8f h0 = vzero, h1 = vzero;
        for (int kb = 0; kb < D_; kb += 32) {
            // stage W1 slab transposed: Bs[n][k], n in [0,64), k in [0,32)
            // thread handles 2 k-rows x 4 n values, packs pairs -> b32 LDS stores
            {
                const int k2 = (tid & 15) * 2;
                const int n0 = (tid >> 4) * 4;
                const unsigned short* g0 =
                    w1bf + ((size_t)e * D_ + (kb + k2)) * H_ + hb + n0;
                const unsigned short* g1 = g0 + H_;
                if (kb + 32 < D_) __builtin_prefetch(g0 + 32 * H_, 0, 1);
                uint2 r0 = *(const uint2*)g0;
                uint2 r1 = *(const uint2*)g1;
                const unsigned short* s0 = (const unsigned short*)&r0;
                const unsigned short* s1 = (const unsigned short*)&r1;
#pragma unroll
                for (int j = 0; j < 4; j++)
                    Bs32[(n0 + j) * 20 + (k2 >> 1)] =
                        (unsigned)s0[j] | ((unsigned)s1[j] << 16);
            }
            __syncthreads();
            // A fragment: rows mi*16..+16, K chunk kb..kb+32 (contiguous k = 16*hi + s, same as B)
            const unsigned short* ap = As + (mi * 16 + hl) * 264 + kb + hi * 16;
            v16bf a = ldfrag2(ap, ap + 8);
            // B fragments: columns (nh*32 .. +16) and (+16 .. +32)
            const unsigned short* bp0 = Bs + (nh * 32 + hl) * 40 + hi * 16;
            const unsigned short* bp1 = Bs + (nh * 32 + 16 + hl) * 40 + hi * 16;
            v16bf bf0 = ldfrag1(bp0);
            v16bf bf1 = ldfrag1(bp1);
            h0 = __builtin_amdgcn_wmma_f32_16x16x32_bf16(false, a, false, bf0,
                                                         (short)0, h0, false, false);
            h1 = __builtin_amdgcn_wmma_f32_16x16x32_bf16(false, a, false, bf1,
                                                         (short)0, h1, false, false);
            __syncthreads();
        }
        // bias + relu -> bf16 -> Hs
        {
            const float bias0 = b1[e * H_ + hb + nh * 32 + hl];
            const float bias1 = b1[e * H_ + hb + nh * 32 + 16 + hl];
#pragma unroll
            for (int j = 0; j < 8; j++) {
                int m = j + hi * 8;                       // C layout: VGPR j -> M=j / j+8
                float v0 = h0[j] + bias0; v0 = v0 > 0.f ? v0 : 0.f;
                float v1 = h1[j] + bias1; v1 = v1 > 0.f ? v1 : 0.f;
                Hs[(mi * 16 + m) * 72 + nh * 32 + hl]      = f2bf(v0);
                Hs[(mi * 16 + m) * 72 + nh * 32 + 16 + hl] = f2bf(v1);
            }
        }
        // ---------------- GEMM2: eo += Hs @ W2[hb:hb+64, :] ----------------
        for (int kb2 = 0; kb2 < 64; kb2 += 32) {
            // stage W2 slab transposed: Bs[n][k], n in [0,256), k in [0,32)
            {
                const int k2 = (tid & 15) * 2;
                const int n0 = (tid >> 4) * 16;
                const unsigned short* g0 =
                    w2bf + ((size_t)e * H_ + hb + kb2 + k2) * D_ + n0;
                const unsigned short* g1 = g0 + D_;
                if (hb + 64 < H_) __builtin_prefetch(g0 + 64 * D_, 0, 1);
                uint4 r0a = *(const uint4*)g0;
                uint4 r0b = *(const uint4*)(g0 + 8);
                uint4 r1a = *(const uint4*)g1;
                uint4 r1b = *(const uint4*)(g1 + 8);
                const unsigned short* s0a = (const unsigned short*)&r0a;
                const unsigned short* s0b = (const unsigned short*)&r0b;
                const unsigned short* s1a = (const unsigned short*)&r1a;
                const unsigned short* s1b = (const unsigned short*)&r1b;
#pragma unroll
                for (int j = 0; j < 8; j++) {
                    Bs32[(n0 + j) * 20 + (k2 >> 1)] =
                        (unsigned)s0a[j] | ((unsigned)s1a[j] << 16);
                    Bs32[(n0 + 8 + j) * 20 + (k2 >> 1)] =
                        (unsigned)s0b[j] | ((unsigned)s1b[j] << 16);
                }
            }
            __syncthreads();
            const unsigned short* ap = Hs + (mi * 16 + hl) * 72 + kb2 + hi * 16;
            v16bf a = ldfrag2(ap, ap + 8);
#pragma unroll
            for (int nj = 0; nj < 8; nj++) {
                const unsigned short* bp =
                    Bs + (nh * 128 + nj * 16 + hl) * 40 + hi * 16;
                v16bf b = ldfrag1(bp);
                eo[nj] = __builtin_amdgcn_wmma_f32_16x16x32_bf16(
                    false, a, false, b, (short)0, eo[nj], false, false);
            }
            __syncthreads();
        }
    }

    // ---------------- epilogue: y[token] = (eo + b2) * prob_max ----------------
    int tok[8]; float pms[8];
#pragma unroll
    for (int j = 0; j < 8; j++) {
        int m = j + hi * 8;
        int slot = tile0 + mi * 16 + m;
        int t = inv_idx[e * CAP + slot];
        tok[j] = t;
        pms[j] = (t >= 0) ? pm[t] : 0.f;
    }
#pragma unroll
    for (int nj = 0; nj < 8; nj++) {
        int n = nh * 128 + nj * 16 + hl;
        float bias = b2[e * D_ + n];
#pragma unroll
        for (int j = 0; j < 8; j++) {
            if (tok[j] >= 0)
                y[(size_t)tok[j] * D_ + n] = (eo[nj][j] + bias) * pms[j];
        }
    }
}

// ---------- launch ----------

extern "C" void kernel_launch(void* const* d_in, const int* in_sizes, int n_in,
                              void* d_out, int out_size, void* d_ws, size_t ws_size,
                              hipStream_t stream) {
    const float* x  = (const float*)d_in[0];   // [B,S,D]
    const float* Wg = (const float*)d_in[1];   // [D,E]
    const float* bg = (const float*)d_in[2];   // [E]
    const float* W1 = (const float*)d_in[3];   // [E,D,H]
    const float* b1 = (const float*)d_in[4];   // [E,H]
    const float* W2 = (const float*)d_in[5];   // [E,H,D]
    const float* b2 = (const float*)d_in[6];   // [E,D]

    float* out = (float*)d_out;
    float* y          = out;                       // [T,D]
    float* out_counts = out + (size_t)T_ * D_;     // [8]
    float* out_psum   = out_counts + 8;            // [8]
    float* out_ndrop  = out_psum + 8;              // [1]
    float* out_pm     = out_ndrop + 1;             // [T]

    char* w = (char*)d_ws;
    int*   routeArr  = (int*)(w + 0);                       // 65536 ints
    int*   rankArr   = (int*)(w + 262144);                  // 65536 ints
    int*   blockhist = (int*)(w + 524288);                  // 2048 ints
    int*   blockoff  = (int*)(w + 532480);                  // 2048 ints
    float* wavepart  = (float*)(w + 540672);                // 16384 floats
    int*   inv_idx   = (int*)(w + 606208);                  // 81920 ints
    unsigned short* bufbf = (unsigned short*)(w + 1048576);   // E*CAP*D bf16 (40MB)
    unsigned short* w1bf  = (unsigned short*)(w + 42991616);  // 2M bf16
    unsigned short* w2bf  = (unsigned short*)(w + 47185920);  // 2M bf16

    moe_prep_kernel<<<1024, 256, 0, stream>>>(W1, W2, (unsigned*)w1bf,
                                              (unsigned*)w2bf, inv_idx);
    moe_router_kernel<<<256, 256, 0, stream>>>(x, Wg, bg, routeArr, rankArr,
                                               blockhist, wavepart, out_pm);
    moe_scan_kernel<<<1, 256, 0, stream>>>(blockhist, wavepart, blockoff,
                                           out_counts, out_psum, out_ndrop);
    moe_dispatch_kernel<<<T_ / 8, 256, 0, stream>>>(x, routeArr, rankArr, blockoff,
                                                    out_pm, bufbf, inv_idx, y);
    moe_expert_kernel<<<E_ * NTILE, 256, 0, stream>>>(bufbf, w1bf, w2bf, b1, b2,
                                                      inv_idx, out_pm, y);
}
